// HivemindAgent_55954833932363
// MI455X (gfx1250) — matrix operand order repaired and checked by reading, not compiled, not run on hardware
//
#include <hip/hip_runtime.h>
#include <hip/hip_bf16.h>

// ---------------------------------------------------------------------------
// Exact dead-code elimination: with Sk==1, softmax over the key axis == 1.0,
// so the cross-attention output is v broadcast => the whole 4-layer stack is
// dead. Live graph: ctx -> (@cWv) -> (@cWo) -> specialist matvecs.
// ---------------------------------------------------------------------------

typedef __attribute__((ext_vector_type(16))) _Float16 v16h;
typedef __attribute__((ext_vector_type(8)))  float    v8f;

#define DMODEL 512
#define BATCH  32

// ctx[b,d] = context[b,0]*ctxW[0,d] + context[b,1]*ctxW[1,d] + ctxb[d]  (K=2, VALU)
__global__ void hm_ctx_kernel(const float* __restrict__ context,   // [B,2]
                              const float* __restrict__ ctxW,      // [2,D]
                              const float* __restrict__ ctxb,      // [D]
                              _Float16* __restrict__ ctx16)        // [B,D] f16 out
{
    int idx = blockIdx.x * blockDim.x + threadIdx.x;   // B*D = 16384
    int b = idx >> 9;
    int d = idx & (DMODEL - 1);
    float v = context[b * 2 + 0] * ctxW[d]
            + context[b * 2 + 1] * ctxW[DMODEL + d]
            + ctxb[d];
    ctx16[idx] = (_Float16)v;
}

// O[32,512] = A(f16)[32,512] @ W(f32->f16)[512,512] + bias, f32 accumulate via
// v_wmma_f32_16x16x32_f16. One wave per 16x16 C tile; 64 tiles -> 64 blocks of
// 32 threads (wave32) so weight streaming spreads across many WGPs.
__global__ void hm_gemm_wmma(const _Float16* __restrict__ A,    // [32,512] row-major f16
                             const float*    __restrict__ W,    // [512,512] row-major f32
                             const float*    __restrict__ bias, // [512]
                             float*          __restrict__ Of32, // [32,512]
                             _Float16*       __restrict__ Of16) // [32,512]
{
    const int lane   = threadIdx.x & 31;
    const int tile   = blockIdx.x;        // 0..63
    const int mt     = tile & 1;          // 2 M tiles (B=32)
    const int nt     = tile >> 1;         // 32 N tiles (D=512)
    const int half_  = lane >> 4;         // lane group 0/1
    const int lane16 = lane & 15;

    const int mBase = mt * 16;
    const int nBase = nt * 16;

    v8f acc = {};

    for (int kk = 0; kk < DMODEL; kk += 32) {
        v16h af, bf;
        // A fragment (ISA 16-bit A 16x32 layout): lane holds M=lane%16,
        // VGPR j covers K = 2*(j%4) + 16*(j/4) + 8*half, two consecutive K.
        #pragma unroll
        for (int j = 0; j < 8; ++j) {
            int k0 = 2 * (j & 3) + 16 * (j >> 2) + 8 * half_;
            const _Float16* ap = A + (mBase + lane16) * DMODEL + kk + k0;
            af[2 * j]     = ap[0];
            af[2 * j + 1] = ap[1];
        }
        // B fragment (rows striped across lanes): element e -> K = 8*half + (e&7) + 16*(e>>3),
        // N = lane%16. Convert f32 weight to f16 on the fly.
        #pragma unroll
        for (int e = 0; e < 16; ++e) {
            int k = 8 * half_ + (e & 7) + 16 * (e >> 3);
            bf[e] = (_Float16)W[(kk + k) * DMODEL + nBase + lane16];
        }
        acc = __builtin_amdgcn_wmma_f32_16x16x32_f16(
            /*neg_a=*/false, af, /*neg_b=*/false, bf,
            /*c_mod=*/(short)0, acc, /*reuse_a=*/false, /*reuse_b=*/false);
    }

    // C/D layout: VGPR i -> M = i + 8*half, N = lane%16
    #pragma unroll
    for (int i = 0; i < 8; ++i) {
        int m = mBase + i + 8 * half_;
        int n = nBase + lane16;
        float v = acc[i] + bias[n];
        Of32[m * DMODEL + n] = v;
        Of16[m * DMODEL + n] = (_Float16)v;
    }
}

// pred[b,o] = last[b,:] . spWp[sid[b]][:,o] + spbp[sid[b]][o]   (and act)
// One block per batch element; 128 threads = 64 pred outputs + 64 act outputs.
__global__ void hm_specialist(const float* __restrict__ last,  // [B,512] f32
                              const int*   __restrict__ sid,   // [B]
                              const float* __restrict__ spWp,  // [NE,512,64]
                              const float* __restrict__ spbp,  // [NE,64]
                              const float* __restrict__ spWa,  // [NE,512,64]
                              const float* __restrict__ spba,  // [NE,64]
                              float*       __restrict__ out)   // pred[B*64] ++ act[B*64]
{
    __shared__ float xs[DMODEL];
    const int b = blockIdx.x;
    const int t = threadIdx.x;            // 0..127
    const int o = t & 63;
    const bool isAct = (t >= 64);
    const int e = sid[b];

    // stage the activation row in LDS (read by all 128 threads)
    for (int d = t; d < DMODEL; d += 128) xs[d] = last[b * DMODEL + d];
    __syncthreads();

    const float* Wm = (isAct ? spWa : spWp) + ((size_t)e * DMODEL) * 64 + o;
    const float  bb = (isAct ? spba : spbp)[e * 64 + o];

    float s = 0.f;
    #pragma unroll 8
    for (int d = 0; d < DMODEL; ++d) s += xs[d] * Wm[(size_t)d * 64];
    s += bb;

    out[(isAct ? BATCH * 64 : 0) + b * 64 + o] = s;
}

extern "C" void kernel_launch(void* const* d_in, const int* in_sizes, int n_in,
                              void* d_out, int out_size, void* d_ws, size_t ws_size,
                              hipStream_t stream) {
    // setup_inputs() order:
    //  0 x_batch  1 context_batch  2 specialist_ids  3 emb
    //  4..11  Wq bq Wk bk Wv bv Wo bo      (dead)
    // 12..15  ln1_s ln1_b ln2_s ln2_b      (dead)
    // 16..19  W1 b1 W2 b2                  (dead)
    // 20 ctxW 21 ctxb
    // 22 cWq 23 cbq 24 cWk 25 cbk          (dead: q/k unused when Sk==1)
    // 26 cWv 27 cbv 28 cWo 29 cbo
    // 30 spWp 31 spbp 32 spWa 33 spba
    const float* context = (const float*)d_in[1];
    const int*   sid     = (const int*)  d_in[2];
    const float* ctxW    = (const float*)d_in[20];
    const float* ctxb    = (const float*)d_in[21];
    const float* cWv     = (const float*)d_in[26];
    const float* cbv     = (const float*)d_in[27];
    const float* cWo     = (const float*)d_in[28];
    const float* cbo     = (const float*)d_in[29];
    const float* spWp    = (const float*)d_in[30];
    const float* spbp    = (const float*)d_in[31];
    const float* spWa    = (const float*)d_in[32];
    const float* spba    = (const float*)d_in[33];

    char* ws = (char*)d_ws;
    _Float16* ctx16  = (_Float16*)(ws);               // 32 KB  [B,D] f16
    _Float16* v16    = (_Float16*)(ws + (32 << 10));  // 32 KB  [B,D] f16
    float*    vf32   = (float*)   (ws + (64 << 10));  // 64 KB  [B,D] f32 (scratch)
    _Float16* last16 = (_Float16*)(ws + (128 << 10)); // 32 KB  (scratch)
    float*    lastf  = (float*)   (ws + (160 << 10)); // 64 KB  [B,D] f32

    // 1) ctx = context @ ctxW + ctxb   (K=2, VALU)
    hm_ctx_kernel<<<64, 256, 0, stream>>>(context, ctxW, ctxb, ctx16);

    // 2) v = ctx @ cWv + cbv           (WMMA f16 -> f32 acc)
    hm_gemm_wmma<<<64, 32, 0, stream>>>(ctx16, cWv, cbv, vf32, v16);

    // 3) last = v @ cWo + cbo          (WMMA f16 -> f32 acc)
    hm_gemm_wmma<<<64, 32, 0, stream>>>(v16, cWo, cbo, lastf, last16);

    // 4) specialist heads -> d_out = [pred (B*64) | act (B*64)], f32
    hm_specialist<<<BATCH, 128, 0, stream>>>(lastf, sid, spWp, spbp, spWa, spba,
                                             (float*)d_out);
}
